// Processor_53618371723681
// MI455X (gfx1250) — matrix operand order repaired and checked by reading, compiled, not run on hardware
//
#include <hip/hip_runtime.h>
#include <stdint.h>

typedef __attribute__((ext_vector_type(16))) __bf16 v16bf;
typedef __attribute__((ext_vector_type(8)))  float  v8f;

union BFrag { uint32_t u[8]; uint4 q[2]; v16bf v; };

// ---- helpers -------------------------------------------------------------

__device__ __forceinline__ uint32_t f2bf_pair(float a, float b) {
    uint32_t ua = __builtin_bit_cast(uint32_t, a);
    uint32_t ub = __builtin_bit_cast(uint32_t, b);
    uint32_t ra = (ua + 0x7FFFu + ((ua >> 16) & 1u)) >> 16;
    uint32_t rb = (ub + 0x7FFFu + ((ub >> 16) & 1u)) & 0xFFFF0000u;
    return ra | rb;
}
__device__ __forceinline__ uint16_t f2bf_bits(float a) {
    uint32_t ua = __builtin_bit_cast(uint32_t, a);
    return (uint16_t)((ua + 0x7FFFu + ((ua >> 16) & 1u)) >> 16);
}
// order-preserving float<->uint encoding for atomic max; enc never produces 0
__device__ __forceinline__ uint32_t enc_ordered(float f) {
    uint32_t b = __builtin_bit_cast(uint32_t, f);
    return (b & 0x80000000u) ? ~b : (b | 0x80000000u);
}
__device__ __forceinline__ float dec_ordered(uint32_t u) {
    if (u == 0u) return 0.0f;   // untouched -> empty segment -> 0
    uint32_t b = (u & 0x80000000u) ? (u & 0x7FFFFFFFu) : ~u;
    return __builtin_bit_cast(float, b);
}

// ---- weight packing: W [Krows x 128] f32 row-major -> bf16 WMMA B-fragments
__global__ void pack_weights(const float* __restrict__ W, uint32_t* __restrict__ out,
                             int Krows, int kTiles) {
    int idx = blockIdx.x * blockDim.x + threadIdx.x;
    int total = 8 * kTiles * 32 * 8;
    if (idx >= total) return;
    int v    = idx & 7;
    int lane = (idx >> 3) & 31;
    int frag = idx >> 8;
    int kT   = frag % kTiles;
    int nT   = frag / kTiles;
    int N     = nT * 16 + (lane & 15);
    int kBase = (lane < 16) ? 0 : 8;
    int K0    = kT * 32 + kBase + 2 * (v & 3) + ((v >= 4) ? 16 : 0);
    float a = (K0     < Krows) ? W[(size_t)K0 * 128 + N]       : 0.0f;
    float b = (K0 + 1 < Krows) ? W[(size_t)(K0 + 1) * 128 + N] : 0.0f;
    out[idx] = f2bf_pair(a, b);
}

__global__ void init_agg(uint32_t* __restrict__ agg, int n) {
    int i = blockIdx.x * blockDim.x + threadIdx.x;
    if (i < n) agg[i] = 0u;
}

#define AT_STRIDE 148   // dwords per LDS row, 144 data (288 bf16 cols) + pad
#define NT_STRIDE 132   // dwords per LDS row, 128 data (256 bf16 cols) + pad
#define H_STRIDE  68    // dwords per hidden row, 64 data (128 bf16) + pad

// ---- edge message kernel: one wave = 32 edges (2 M-tiles share B frags) --
__global__ __launch_bounds__(32, 4)
void edge_msg_kernel(const float* __restrict__ h, const float* __restrict__ w,
                     const int* __restrict__ src, const int* __restrict__ dst,
                     const uint32_t* __restrict__ W1p, const float* __restrict__ b1,
                     const uint32_t* __restrict__ W2p, const float* __restrict__ b2,
                     uint32_t* __restrict__ aggEnc, int E) {
    __shared__ __align__(16) uint32_t lds[32 * AT_STRIDE];   // 18944 B
    uint32_t* ldsH = lds;   // hidden tile overlays A tile (A dead after layer 1)
    const int lane  = threadIdx.x;
    const int halfN = lane & 15;
    const int hi    = lane >> 4;
    const int e0    = blockIdx.x * 32;
    const v8f vzero = {};

    // --- build 32 x [h_src(128) | h_dst(128) | w | pad] bf16 tile in LDS
    for (int e = 0; e < 32; ++e) {
        int eg = e0 + e; if (eg > E - 1) eg = E - 1;
        int sN = src[eg], dN = dst[eg];
        float4 hs = *(const float4*)(h + (size_t)sN * 128 + lane * 4);
        float4 hd = *(const float4*)(h + (size_t)dN * 128 + lane * 4);
        uint32_t* row = lds + e * AT_STRIDE;
        row[2 * lane]          = f2bf_pair(hs.x, hs.y);
        row[2 * lane + 1]      = f2bf_pair(hs.z, hs.w);
        row[64 + 2 * lane]     = f2bf_pair(hd.x, hd.y);
        row[64 + 2 * lane + 1] = f2bf_pair(hd.z, hd.w);
        if (lane < 16) row[128 + lane] = (lane == 0) ? f2bf_pair(w[eg], 0.0f) : 0u;
    }
    __syncthreads();

    // --- layer 1: [32x288] @ [288x128]; C=0 on first k-tile, bias in epilogue
    v8f acc[2][8];
    const uint32_t* aRow0 = lds + (halfN)      * AT_STRIDE;
    const uint32_t* aRow1 = lds + (16 + halfN) * AT_STRIDE;
    #pragma unroll
    for (int kT = 0; kT < 9; ++kT) {
        int p0 = kT * 16 + hi * 4;
        BFrag A0, A1;
        A0.q[0] = *(const uint4*)(aRow0 + p0);
        A0.q[1] = *(const uint4*)(aRow0 + p0 + 8);
        A1.q[0] = *(const uint4*)(aRow1 + p0);
        A1.q[1] = *(const uint4*)(aRow1 + p0 + 8);
        BFrag Bb[2];
        {   const uint32_t* bp = W1p + (size_t)((0 * 9 + kT) * 32 + lane) * 8;
            Bb[0].q[0] = *(const uint4*)(bp);
            Bb[0].q[1] = *(const uint4*)(bp + 4); }
        #pragma unroll
        for (int nT = 0; nT < 8; ++nT) {
            if (nT < 7) {   // prefetch next B fragment (double buffer)
                const uint32_t* bp = W1p + (size_t)(((nT + 1) * 9 + kT) * 32 + lane) * 8;
                Bb[(nT + 1) & 1].q[0] = *(const uint4*)(bp);
                Bb[(nT + 1) & 1].q[1] = *(const uint4*)(bp + 4);
            }
            const v16bf Bv = Bb[nT & 1].v;
            acc[0][nT] = __builtin_amdgcn_wmma_f32_16x16x32_bf16(
                false, A0.v, false, Bv, (short)0, kT ? acc[0][nT] : vzero, false, false);
            acc[1][nT] = __builtin_amdgcn_wmma_f32_16x16x32_bf16(
                false, A1.v, false, Bv, (short)0, kT ? acc[1][nT] : vzero, false, false);
        }
    }
    __syncthreads();   // A tile reads complete before overlaying hidden tile

    // --- bias + ReLU, transpose to A-layout via LDS (bf16)
    #pragma unroll
    for (int nT = 0; nT < 8; ++nT) {
        int col = nT * 16 + halfN;
        float bv = b1[col];
        #pragma unroll
        for (int t = 0; t < 2; ++t) {
            #pragma unroll
            for (int r = 0; r < 8; ++r) {
                float x = acc[t][nT][r] + bv; x = x > 0.0f ? x : 0.0f;
                int mm = t * 16 + r + hi * 8;
                ((uint16_t*)(ldsH + mm * H_STRIDE))[col] = f2bf_bits(x);
            }
        }
    }
    __syncthreads();

    // --- layer 2: [32x128] @ [128x128]
    v8f acc2[2][8];
    const uint32_t* hRow0 = ldsH + (halfN)      * H_STRIDE;
    const uint32_t* hRow1 = ldsH + (16 + halfN) * H_STRIDE;
    #pragma unroll
    for (int kT = 0; kT < 4; ++kT) {
        int p0 = kT * 16 + hi * 4;
        BFrag A0, A1;
        A0.q[0] = *(const uint4*)(hRow0 + p0);
        A0.q[1] = *(const uint4*)(hRow0 + p0 + 8);
        A1.q[0] = *(const uint4*)(hRow1 + p0);
        A1.q[1] = *(const uint4*)(hRow1 + p0 + 8);
        BFrag Bb[2];
        {   const uint32_t* bp = W2p + (size_t)((0 * 4 + kT) * 32 + lane) * 8;
            Bb[0].q[0] = *(const uint4*)(bp);
            Bb[0].q[1] = *(const uint4*)(bp + 4); }
        #pragma unroll
        for (int nT = 0; nT < 8; ++nT) {
            if (nT < 7) {
                const uint32_t* bp = W2p + (size_t)(((nT + 1) * 4 + kT) * 32 + lane) * 8;
                Bb[(nT + 1) & 1].q[0] = *(const uint4*)(bp);
                Bb[(nT + 1) & 1].q[1] = *(const uint4*)(bp + 4);
            }
            const v16bf Bv = Bb[nT & 1].v;
            acc2[0][nT] = __builtin_amdgcn_wmma_f32_16x16x32_bf16(
                false, A0.v, false, Bv, (short)0, kT ? acc2[0][nT] : vzero, false, false);
            acc2[1][nT] = __builtin_amdgcn_wmma_f32_16x16x32_bf16(
                false, A1.v, false, Bv, (short)0, kT ? acc2[1][nT] : vzero, false, false);
        }
    }

    // --- bias + scatter-max onto destination rows (L2-resident atomics)
    int dstRow[2][8];
    #pragma unroll
    for (int t = 0; t < 2; ++t)
        #pragma unroll
        for (int r = 0; r < 8; ++r) {
            int eg = e0 + t * 16 + r + hi * 8;
            dstRow[t][r] = (eg < E) ? dst[eg] : -1;
        }
    #pragma unroll
    for (int nT = 0; nT < 8; ++nT) {
        int col = nT * 16 + halfN;
        float bv = b2[col];
        #pragma unroll
        for (int t = 0; t < 2; ++t)
            #pragma unroll
            for (int r = 0; r < 8; ++r)
                if (dstRow[t][r] >= 0)
                    atomicMax(aggEnc + (size_t)dstRow[t][r] * 128 + col,
                              enc_ordered(acc2[t][nT][r] + bv));
    }
}

// ---- node update kernel: one wave = 32 nodes ----------------------------
__global__ __launch_bounds__(32, 4)
void node_upd_kernel(const float* __restrict__ h,
                     const uint32_t* __restrict__ U1p, const float* __restrict__ c1,
                     const uint32_t* __restrict__ U2p, const float* __restrict__ c2,
                     float* __restrict__ out, const uint32_t* __restrict__ aggEnc, int N) {
    __shared__ __align__(16) uint32_t lds[32 * NT_STRIDE];   // 16896 B
    uint32_t* ldsH = lds;
    const int lane  = threadIdx.x;
    const int halfN = lane & 15;
    const int hi    = lane >> 4;
    const int n0    = blockIdx.x * 32;
    const v8f vzero = {};

    // --- build 32 x [h(128) | agg(128)] bf16 tile
    for (int nd = 0; nd < 32; ++nd) {
        int ng = n0 + nd; if (ng > N - 1) ng = N - 1;
        float4 hv = *(const float4*)(h + (size_t)ng * 128 + lane * 4);
        uint4  av = *(const uint4*)(aggEnc + (size_t)ng * 128 + lane * 4);
        uint32_t* row = lds + nd * NT_STRIDE;
        row[2 * lane]          = f2bf_pair(hv.x, hv.y);
        row[2 * lane + 1]      = f2bf_pair(hv.z, hv.w);
        row[64 + 2 * lane]     = f2bf_pair(dec_ordered(av.x), dec_ordered(av.y));
        row[64 + 2 * lane + 1] = f2bf_pair(dec_ordered(av.z), dec_ordered(av.w));
    }
    __syncthreads();

    // --- layer 1: [32x256] @ [256x128]
    v8f acc[2][8];
    const uint32_t* aRow0 = lds + (halfN)      * NT_STRIDE;
    const uint32_t* aRow1 = lds + (16 + halfN) * NT_STRIDE;
    #pragma unroll
    for (int kT = 0; kT < 8; ++kT) {
        int p0 = kT * 16 + hi * 4;
        BFrag A0, A1;
        A0.q[0] = *(const uint4*)(aRow0 + p0);
        A0.q[1] = *(const uint4*)(aRow0 + p0 + 8);
        A1.q[0] = *(const uint4*)(aRow1 + p0);
        A1.q[1] = *(const uint4*)(aRow1 + p0 + 8);
        BFrag Bb[2];
        {   const uint32_t* bp = U1p + (size_t)((0 * 8 + kT) * 32 + lane) * 8;
            Bb[0].q[0] = *(const uint4*)(bp);
            Bb[0].q[1] = *(const uint4*)(bp + 4); }
        #pragma unroll
        for (int nT = 0; nT < 8; ++nT) {
            if (nT < 7) {
                const uint32_t* bp = U1p + (size_t)(((nT + 1) * 8 + kT) * 32 + lane) * 8;
                Bb[(nT + 1) & 1].q[0] = *(const uint4*)(bp);
                Bb[(nT + 1) & 1].q[1] = *(const uint4*)(bp + 4);
            }
            const v16bf Bv = Bb[nT & 1].v;
            acc[0][nT] = __builtin_amdgcn_wmma_f32_16x16x32_bf16(
                false, A0.v, false, Bv, (short)0, kT ? acc[0][nT] : vzero, false, false);
            acc[1][nT] = __builtin_amdgcn_wmma_f32_16x16x32_bf16(
                false, A1.v, false, Bv, (short)0, kT ? acc[1][nT] : vzero, false, false);
        }
    }
    __syncthreads();

    // --- bias + ReLU -> hidden LDS (overlaid)
    #pragma unroll
    for (int nT = 0; nT < 8; ++nT) {
        int col = nT * 16 + halfN;
        float bv = c1[col];
        #pragma unroll
        for (int t = 0; t < 2; ++t)
            #pragma unroll
            for (int r = 0; r < 8; ++r) {
                float x = acc[t][nT][r] + bv; x = x > 0.0f ? x : 0.0f;
                int mm = t * 16 + r + hi * 8;
                ((uint16_t*)(ldsH + mm * H_STRIDE))[col] = f2bf_bits(x);
            }
    }
    __syncthreads();

    // --- layer 2: [32x128] @ [128x128]
    v8f acc2[2][8];
    const uint32_t* hRow0 = ldsH + (halfN)      * H_STRIDE;
    const uint32_t* hRow1 = ldsH + (16 + halfN) * H_STRIDE;
    #pragma unroll
    for (int kT = 0; kT < 4; ++kT) {
        int p0 = kT * 16 + hi * 4;
        BFrag A0, A1;
        A0.q[0] = *(const uint4*)(hRow0 + p0);
        A0.q[1] = *(const uint4*)(hRow0 + p0 + 8);
        A1.q[0] = *(const uint4*)(hRow1 + p0);
        A1.q[1] = *(const uint4*)(hRow1 + p0 + 8);
        BFrag Bb[2];
        {   const uint32_t* bp = U2p + (size_t)((0 * 4 + kT) * 32 + lane) * 8;
            Bb[0].q[0] = *(const uint4*)(bp);
            Bb[0].q[1] = *(const uint4*)(bp + 4); }
        #pragma unroll
        for (int nT = 0; nT < 8; ++nT) {
            if (nT < 7) {
                const uint32_t* bp = U2p + (size_t)(((nT + 1) * 4 + kT) * 32 + lane) * 8;
                Bb[(nT + 1) & 1].q[0] = *(const uint4*)(bp);
                Bb[(nT + 1) & 1].q[1] = *(const uint4*)(bp + 4);
            }
            const v16bf Bv = Bb[nT & 1].v;
            acc2[0][nT] = __builtin_amdgcn_wmma_f32_16x16x32_bf16(
                false, A0.v, false, Bv, (short)0, kT ? acc2[0][nT] : vzero, false, false);
            acc2[1][nT] = __builtin_amdgcn_wmma_f32_16x16x32_bf16(
                false, A1.v, false, Bv, (short)0, kT ? acc2[1][nT] : vzero, false, false);
        }
    }

    // --- bias + store final output
    #pragma unroll
    for (int nT = 0; nT < 8; ++nT) {
        int col = nT * 16 + halfN;
        float bv = c2[col];
        #pragma unroll
        for (int t = 0; t < 2; ++t)
            #pragma unroll
            for (int r = 0; r < 8; ++r) {
                int ng = n0 + t * 16 + r + hi * 8;
                if (ng < N) out[(size_t)ng * 128 + col] = acc2[t][nT][r] + bv;
            }
    }
}

// ---- host launcher -------------------------------------------------------
extern "C" void kernel_launch(void* const* d_in, const int* in_sizes, int n_in,
                              void* d_out, int out_size, void* d_ws, size_t ws_size,
                              hipStream_t stream) {
    const float* h  = (const float*)d_in[0];
    const float* w  = (const float*)d_in[1];
    const int*  src = (const int*)d_in[2];
    const int*  dst = (const int*)d_in[3];
    const float* W1 = (const float*)d_in[4];
    const float* b1 = (const float*)d_in[5];
    const float* W2 = (const float*)d_in[6];
    const float* b2 = (const float*)d_in[7];
    const float* U1 = (const float*)d_in[8];
    const float* c1 = (const float*)d_in[9];
    const float* U2 = (const float*)d_in[10];
    const float* c2 = (const float*)d_in[11];

    const int N = in_sizes[0] / 128;
    const int E = in_sizes[1];

    uint32_t* ws  = (uint32_t*)d_ws;
    uint32_t* W1p = ws;              // 8 nT * 9 kT * 256 = 18432 dwords
    uint32_t* W2p = ws + 18432;      // 8 nT * 4 kT * 256 =  8192 dwords
    uint32_t* U1p = ws + 26624;      // 8 nT * 8 kT * 256 = 16384 dwords
    uint32_t* U2p = ws + 43008;      // 8 nT * 4 kT * 256 =  8192 dwords

    pack_weights<<<dim3((18432 + 255) / 256), dim3(256), 0, stream>>>(W1, W1p, 257, 9);
    pack_weights<<<dim3((8192  + 255) / 256), dim3(256), 0, stream>>>(W2, W2p, 128, 4);
    pack_weights<<<dim3((16384 + 255) / 256), dim3(256), 0, stream>>>(U1, U1p, 256, 8);
    pack_weights<<<dim3((8192  + 255) / 256), dim3(256), 0, stream>>>(U2, U2p, 128, 4);

    init_agg<<<dim3((N * 128 + 255) / 256), dim3(256), 0, stream>>>((uint32_t*)d_out, N * 128);

    edge_msg_kernel<<<dim3((E + 31) / 32), dim3(32), 0, stream>>>(
        h, w, src, dst, W1p, b1, W2p, b2, (uint32_t*)d_out, E);

    node_upd_kernel<<<dim3((N + 31) / 32), dim3(32), 0, stream>>>(
        h, U1p, c1, U2p, c2, (float*)d_out, (const uint32_t*)d_out, N);
}
